// MultiHeadAttention2D_11098195493208
// MI455X (gfx1250) — matrix operand order repaired and checked by reading, not compiled, not run on hardware
//
#include <hip/hip_runtime.h>

// MultiHeadAttention2D for MI455X (gfx1250), wave32 + WMMA + TDM.
// B=8, H=W=32 (N=1024), C=256, HEADS=8, D=HID=32.
// Kernel 1: fused QKV projection GEMM (f32->f16, WMMA f16, f32 accum),
//           epilogue applies bias, K-scale (1/sqrt(32)) and tanh(V); stores
//           f16 Q/K/V to workspace in [b*8+h][n][d] layout (12 MB total).
// Kernel 2: flash attention per (b,h): 128 query rows per block (8 waves x 16
//           rows). K tiles are streamed into LDS by the Tensor Data Mover
//           (tensor_load_to_lds, TENSORcnt), V tiles are transposed in
//           registers, online softmax uses wave32 shfl reductions, P is staged
//           via per-wave LDS (LDS ops are in-order within a wave on CDNA5),
//           and the PV product accumulates via WMMA.

typedef __attribute__((ext_vector_type(16))) _Float16 v16h;
typedef __attribute__((ext_vector_type(8)))  _Float16 v8h;
typedef __attribute__((ext_vector_type(8)))  float    v8f;
typedef __attribute__((ext_vector_type(4)))  unsigned int u32x4;
typedef __attribute__((ext_vector_type(8)))  int i32x8;
typedef __attribute__((ext_vector_type(4)))  int i32x4;

#define NB    8
#define NHEAD 8
#define NSEQ  1024
#define CDIM  256
#define DDIM  32
#define BM    128
#define KT    256

static __device__ __forceinline__ v16h cat8(v8h lo, v8h hi) {
  return __builtin_shufflevector(lo, hi, 0,1,2,3,4,5,6,7,8,9,10,11,12,13,14,15);
}

static __device__ __forceinline__ v8f wmma_f16(v16h a, v16h b, v8f c) {
  // D = A(16x32 f16) * B(32x16 f16) + C(16x16 f32)
  return __builtin_amdgcn_wmma_f32_16x16x32_f16(false, a, false, b,
                                                (short)0, c, false, false);
}

// ---------------------------------------------------------------------------
// Kernel 1: QKV projection. grid = (8192/128, HEADS), block = 256 (8 waves).
// Each wave computes a 16-row x 96-col (q|k|v) output slab via WMMA.
// ---------------------------------------------------------------------------
__global__ __launch_bounds__(256)
void mha2d_qkv_proj(const float* __restrict__ seq,
                    const float* __restrict__ Wq, const float* __restrict__ Wk,
                    const float* __restrict__ Wv,
                    const float* __restrict__ bq, const float* __restrict__ bk,
                    const float* __restrict__ bv,
                    _Float16* __restrict__ qws, _Float16* __restrict__ kws,
                    _Float16* __restrict__ vws)
{
  // Transposed weights (d-major, c contiguous) in f16: B^T rows are contiguous
  // so B-fragments are single 32-byte LDS reads.
  __shared__ __align__(32) _Float16 WT[3][DDIM][CDIM];

  const int h   = blockIdx.y;
  const int tid = threadIdx.x;

#pragma unroll
  for (int m = 0; m < 3; ++m) {
    const float* W = (m == 0 ? Wq : (m == 1 ? Wk : Wv)) + (size_t)h * CDIM * DDIM;
    for (int e = tid; e < DDIM * CDIM; e += 256) {
      const int d = e >> 8;          // e / 256
      const int c = e & (CDIM - 1);  // e % 256
      WT[m][d][c] = (_Float16)W[c * DDIM + d];
    }
  }
  __syncthreads();

  const int wave = tid >> 5, lane = tid & 31;
  const int rlo = lane & 15;   // row (A) / column (B,D) within tile
  const int hs  = lane >> 4;   // lane-half select
  const int arow = blockIdx.x * BM + wave * 16 + rlo;
  const float* srow = seq + (size_t)arow * CDIM;

  v8f acc[3][2];
#pragma unroll
  for (int m = 0; m < 3; ++m)
#pragma unroll
    for (int t = 0; t < 2; ++t)
#pragma unroll
      for (int r = 0; r < 8; ++r) acc[m][t][r] = 0.0f;

#pragma unroll
  for (int ks = 0; ks < CDIM / 32; ++ks) {
    // A fragment: 16 f32 -> f16 per lane (K = hs*16 .. hs*16+15).
    const int cbase = ks * 32 + hs * 16;
    v16h afrag;
#pragma unroll
    for (int i = 0; i < 8; ++i) {
      afrag[i]     = (_Float16)srow[cbase + i];
      afrag[8 + i] = (_Float16)srow[cbase + 8 + i];
    }
#pragma unroll
    for (int m = 0; m < 3; ++m)
#pragma unroll
      for (int t = 0; t < 2; ++t) {
        // B fragment: lane n holds column d = t*16+n, K = ks*32 + hs*16 .. +15.
        const v16h bfrag =
            *(const v16h*)&WT[m][t * 16 + rlo][ks * 32 + hs * 16];
        acc[m][t] = wmma_f16(afrag, bfrag, acc[m][t]);
      }
  }

  const float scale = 0.17677669529663687f; // 1/sqrt(32)
  const int rowbase = blockIdx.x * BM + wave * 16 + hs * 8; // D-layout rows
#pragma unroll
  for (int t = 0; t < 2; ++t) {
    const int d = t * 16 + rlo;
    const float bqv = bq[h * DDIM + d];
    const float bkv = bk[h * DDIM + d];
    const float bvv = bv[h * DDIM + d];
#pragma unroll
    for (int r = 0; r < 8; ++r) {
      const int grow = rowbase + r;
      const int b = grow >> 10, n = grow & (NSEQ - 1);
      const size_t o = ((size_t)((b * NHEAD + h) * NSEQ + n)) * DDIM + d;
      qws[o] = (_Float16)(acc[0][t][r] + bqv);
      kws[o] = (_Float16)((acc[1][t][r] + bkv) * scale);
      vws[o] = (_Float16)tanhf(acc[2][t][r] + bvv);
    }
  }
}

// ---------------------------------------------------------------------------
// Kernel 2: flash attention. grid = (NSEQ/BM, B*HEADS), block = 256 (8 waves).
// ---------------------------------------------------------------------------
__global__ __launch_bounds__(256)
void mha2d_flash_attn(const _Float16* __restrict__ qws,
                      const _Float16* __restrict__ kws,
                      const _Float16* __restrict__ vws,
                      float* __restrict__ out)
{
  __shared__ __align__(32) _Float16 Kt[KT][DDIM];    // [key][d]  row-major
  __shared__ __align__(32) _Float16 Vt[DDIM][KT];    // [d][key]  transposed
  __shared__ __align__(32) _Float16 Pst[8][16][40];  // per-wave P staging

  const int bh = blockIdx.y, b = bh >> 3, h = bh & 7;
  const int mtile = blockIdx.x;
  const int tid = threadIdx.x, wave = tid >> 5, lane = tid & 31;
  const int rlo = lane & 15, hs = lane >> 4;

  const _Float16* qb = qws + (size_t)bh * NSEQ * DDIM;
  const _Float16* kb = kws + (size_t)bh * NSEQ * DDIM;
  const _Float16* vb = vws + (size_t)bh * NSEQ * DDIM;

  // Q A-fragment (rows = this wave's 16 query rows, K = D = 32).
  const int qrow = mtile * BM + wave * 16 + rlo;
  v16h qf;
  {
    const v8h* qp = (const v8h*)(qb + (size_t)qrow * DDIM + hs * 16);
    qf = cat8(qp[0], qp[1]); // d0..d0+15
  }

  v8f a0, a1;            // output accumulators, d = 0..15 and 16..31
  float mrow[8], lrow[8];
#pragma unroll
  for (int r = 0; r < 8; ++r) {
    a0[r] = 0.0f; a1[r] = 0.0f; mrow[r] = -1e30f; lrow[r] = 0.0f;
  }

  for (int jt = 0; jt < NSEQ / KT; ++jt) {
    __syncthreads();
#if __has_builtin(__builtin_amdgcn_tensor_load_to_lds) && \
    __has_builtin(__builtin_amdgcn_s_wait_tensorcnt)
    // K tile via Tensor Data Mover: one wave issues a 1-D DMA of 2048 u64
    // elements (16 KB) into Kt; tracked with TENSORcnt. D# packing per
    // CDNA5 ISA ch.8 (group0: count/lds_addr/global_addr/type=2; group1:
    // data_size=8B, tensor_dim0=tile_dim0=2048, stride0=2048).
    if (wave == 0) {
      const unsigned long long ga =
          (unsigned long long)(size_t)(const void*)(kb + (size_t)jt * KT * DDIM);
      const unsigned ldsoff = (unsigned)(size_t)(void*)&Kt[0][0]; // addr[31:0] = LDS offset
      const unsigned tdim0 = (KT * DDIM) / 4;  // 2048 8-byte elements
      u32x4 g0;
      g0[0] = 1u;                                          // count=1 (valid D#)
      g0[1] = ldsoff;                                      // lds_addr (bytes)
      g0[2] = (unsigned)(ga & 0xffffffffull);              // global_addr[31:0]
      g0[3] = (unsigned)((ga >> 32) & 0x01ffffffull)       // global_addr[56:32]
              | (2u << 30);                                // type=2 ("image")
      i32x8 g1;
      g1[0] = (int)(3u << 16);                 // wg_mask=0, data_size=3 (8B)
      g1[1] = (int)((tdim0 & 0xffffu) << 16);  // tensor_dim0[15:0] @ bits 63:48
      g1[2] = (int)((tdim0 >> 16) | (1u << 16)); // tensor_dim0[31:16] | tensor_dim1=1
      g1[3] = (int)(tdim0 << 16);              // tensor_dim1[31:16]=0 | tile_dim0
      g1[4] = (int)1u;                         // tile_dim1=1, tile_dim2=0
      g1[5] = (int)tdim0;                      // tensor_dim0_stride[31:0]
      g1[6] = 0;                               // stride0[47:32], stride1[15:0]
      g1[7] = 0;                               // stride1[47:16]
      i32x4 z4;
      z4[0] = 0; z4[1] = 0; z4[2] = 0; z4[3] = 0; // groups 2/3: dims unused
      i32x8 z8;
#pragma unroll
      for (int i = 0; i < 8; ++i) z8[i] = 0;
      // 6-arg form (clang-23 / therock-10.0 headers): groups 0..3 + extra + cpol
      __builtin_amdgcn_tensor_load_to_lds(g0, g1, z4, z4, z8, 0);
    }
#else
    { // Fallback: flat 128-bit copy (layout in LDS == layout in memory).
      const uint4* src = (const uint4*)(kb + (size_t)jt * KT * DDIM);
      uint4* dst = (uint4*)&Kt[0][0];
#pragma unroll
      for (int i = 0; i < (KT * DDIM) / (8 * 256); ++i)
        dst[tid + i * 256] = src[tid + i * 256];
    }
#endif
    { // V tile: transpose [key][d] -> [d][key] while loading.
      const unsigned int* src = (const unsigned int*)(vb + (size_t)jt * KT * DDIM);
#pragma unroll
      for (int i = 0; i < (KT * DDIM / 2) / 256; ++i) {
        const int u = tid + i * 256;
        const int key = u >> 4, dp = (u & 15) * 2;
        union { unsigned int w; _Float16 f[2]; } cv;
        cv.w = src[u];
        Vt[dp][key]     = cv.f[0];
        Vt[dp + 1][key] = cv.f[1];
      }
    }
    if (jt + 1 < NSEQ / KT) { // prefetch next tile -> global_prefetch_b8
      __builtin_prefetch((const char*)(kb + (size_t)(jt + 1) * KT * DDIM) + tid * 64, 0, 0);
      __builtin_prefetch((const char*)(vb + (size_t)(jt + 1) * KT * DDIM) + tid * 64, 0, 0);
    }
#if __has_builtin(__builtin_amdgcn_tensor_load_to_lds) && \
    __has_builtin(__builtin_amdgcn_s_wait_tensorcnt)
    if (wave == 0) __builtin_amdgcn_s_wait_tensorcnt(0); // drain TDM before barrier
#endif
    __syncthreads();

    for (int js = 0; js < KT / 32; ++js) {
      const int kb0 = js * 32;
      // S = Q * K^T for 32 keys: two 16x16 f32 tiles. K rows are B^T rows.
      const v16h kf0 = *(const v16h*)&Kt[kb0 + rlo][hs * 16];
      const v16h kf1 = *(const v16h*)&Kt[kb0 + 16 + rlo][hs * 16];
      v8f z8;
#pragma unroll
      for (int r = 0; r < 8; ++r) z8[r] = 0.0f;
      v8f s0 = wmma_f16(qf, kf0, z8);
      v8f s1 = wmma_f16(qf, kf1, z8);

      // Online softmax: row stats live replicated across each 16-lane half.
      float e0[8], e1[8];
#pragma unroll
      for (int r = 0; r < 8; ++r) {
        float mx = fmaxf(s0[r], s1[r]);
        mx = fmaxf(mx, __shfl_xor(mx, 1, 32));
        mx = fmaxf(mx, __shfl_xor(mx, 2, 32));
        mx = fmaxf(mx, __shfl_xor(mx, 4, 32));
        mx = fmaxf(mx, __shfl_xor(mx, 8, 32));
        const float mn = fmaxf(mrow[r], mx);
        const float cr = __expf(mrow[r] - mn);
        const float p0 = __expf(s0[r] - mn);
        const float p1 = __expf(s1[r] - mn);
        float rs = p0 + p1;
        rs += __shfl_xor(rs, 1, 32);
        rs += __shfl_xor(rs, 2, 32);
        rs += __shfl_xor(rs, 4, 32);
        rs += __shfl_xor(rs, 8, 32);
        mrow[r] = mn;
        lrow[r] = lrow[r] * cr + rs;
        a0[r] *= cr;
        a1[r] *= cr;
        e0[r] = p0;
        e1[r] = p1;
      }
      // Stage P (f16) through per-wave LDS to convert D-layout -> A-layout.
#pragma unroll
      for (int r = 0; r < 8; ++r) {
        const int pr = r + hs * 8;
        Pst[wave][pr][rlo]      = (_Float16)e0[r];
        Pst[wave][pr][16 + rlo] = (_Float16)e1[r];
      }
      // LDS ops are in-order within a wave on CDNA5; this pins compiler order
      // and drains DScnt before the cross-lane fragment reads below.
      asm volatile("s_wait_dscnt 0x0" ::: "memory");
      v16h pf;
      {
        const _Float16* pp = &Pst[wave][rlo][hs * 16];
        pf = cat8(*(const v8h*)pp, *(const v8h*)(pp + 8));
      }
      // V B-fragments: B^T rows are Vt rows (d-major), keys contiguous.
      const v16h vf0 = *(const v16h*)&Vt[rlo][kb0 + hs * 16];
      const v16h vf1 = *(const v16h*)&Vt[16 + rlo][kb0 + hs * 16];
      a0 = wmma_f16(pf, vf0, a0);
      a1 = wmma_f16(pf, vf1, a1);
      asm volatile("" ::: "memory"); // keep next iter's P stores after these reads
    }
  }

  // Epilogue: normalize and scatter to [b, n, head*32 + d] (f32, coalesced).
  const int nbase = mtile * BM + wave * 16 + hs * 8;
#pragma unroll
  for (int r = 0; r < 8; ++r) {
    const float inv = 1.0f / lrow[r];
    const size_t o = ((size_t)(b * NSEQ + nbase + r)) * (NHEAD * DDIM) + h * DDIM;
    out[o + rlo]      = a0[r] * inv;
    out[o + 16 + rlo] = a1[r] * inv;
  }
}

// ---------------------------------------------------------------------------
extern "C" void kernel_launch(void* const* d_in, const int* in_sizes, int n_in,
                              void* d_out, int out_size, void* d_ws, size_t ws_size,
                              hipStream_t stream) {
  (void)in_sizes; (void)n_in; (void)out_size; (void)ws_size;
  const float* seq = (const float*)d_in[0];
  // d_in[1] = ctx (unused by the reference computation)
  const float* Wq = (const float*)d_in[2];
  const float* Wk = (const float*)d_in[3];
  const float* Wv = (const float*)d_in[4];
  const float* bq = (const float*)d_in[5];
  const float* bk = (const float*)d_in[6];
  const float* bv = (const float*)d_in[7];
  float* out = (float*)d_out;

  // Workspace: q|k|v in f16, each B*HEADS*N*D = 2M elements (4 MB), 12 MB total.
  _Float16* qws = (_Float16*)d_ws;
  const size_t per = (size_t)NB * NHEAD * NSEQ * DDIM;
  _Float16* kws = qws + per;
  _Float16* vws = kws + per;

  mha2d_qkv_proj<<<dim3((NB * NSEQ) / BM, NHEAD), 256, 0, stream>>>(
      seq, Wq, Wk, Wv, bq, bk, bv, qws, kws, vws);
  mha2d_flash_attn<<<dim3(NSEQ / BM, NB * NHEAD), 256, 0, stream>>>(
      qws, kws, vws, out);
}
